// GraphAttentionLayer_64587718197779
// MI455X (gfx1250) — matrix operand order repaired and checked by reading, not compiled, not run on hardware
//
#include <hip/hip_runtime.h>
#include <hip/hip_bf16.h>

// GAT layer for gfx1250 (MI455X), wave32 + WMMA + async global->LDS staging.
//
// Math: softmax_j((s_i + s_j)*scale + mask) == softmax_j(s_j*scale + mask)
// since s_i is constant along j. With u[j,h] = exp(s_j[j,h]*scale):
//   out[i,h,:] = (adj @ (u .* xp))[i,h,:] / (adj @ u)[i,h]
// The whole attention collapses to one dense GEMM  Raw = adj(4096x4096) @ V(4096x144)
// where V cols 0..127 = u[h]*xp[h,d], cols 128..131 = u[h], cols 132..143 = 0 pad.
// adj is exactly representable in fp16 -> v_wmma_f32_16x16x32_f16 with f32 accum.

#define NN      4096
#define IN_DIM  128
#define OC      128     // OUT_DIM * HEADS
#define NCOL    144     // 128 value cols + 4 denom cols + 12 pad (9 N-tiles of 16)
#define NT      9
#define BM      32      // rows per block (2 waves x 16)
#define BK      32      // K per WMMA step
#define VPAD    40      // LDS row stride in halfs (80B: 16B aligned, conflict-free)

typedef __attribute__((ext_vector_type(16))) _Float16 v16h;
typedef __attribute__((ext_vector_type(8)))  _Float16 v8h;
typedef __attribute__((ext_vector_type(8)))  float    v8f;
typedef __attribute__((ext_vector_type(4)))  float    v4f;

// -------- CDNA5 async global->LDS copy (ASYNCcnt-tracked, no VGPR bounce) ----
__device__ __forceinline__ void async_cp16(unsigned lds_addr, const void* gaddr) {
    // LDS[lds_addr] = MEM[gaddr], 16 bytes per lane, tracked with ASYNCcnt.
    asm volatile("global_load_async_to_lds_b128 %0, %1, off"
                 :: "v"(lds_addr), "v"(gaddr)
                 : "memory");
}
__device__ __forceinline__ void wait_async_le9() {
#if __has_builtin(__builtin_amdgcn_s_wait_asynccnt)
    __builtin_amdgcn_s_wait_asynccnt(9);
#else
    asm volatile("s_wait_asynccnt 0x9" ::: "memory");
#endif
}
__device__ __forceinline__ void wait_async_le0() {
#if __has_builtin(__builtin_amdgcn_s_wait_asynccnt)
    __builtin_amdgcn_s_wait_asynccnt(0);
#else
    asm volatile("s_wait_asynccnt 0x0" ::: "memory");
#endif
}

// ---------------------------------------------------------------------------
// Kernel 1: xp = x@W, per-head score s_j, u = exp(s_j*scale), write Vt fp16.
// Vt stored COLUMN-major: Vt[c*4096 + j] so kernel-2 B-tiles are contiguous.
// One block per node; 128 threads = 4 waves; each wave32 == one head.
// ---------------------------------------------------------------------------
__global__ void gat_xp_v_kernel(const float* __restrict__ x,
                                const float* __restrict__ W,
                                const float* __restrict__ a,
                                _Float16* __restrict__ Vt) {
    __shared__ float xs[IN_DIM];
    const int node = blockIdx.x;
    const int t    = threadIdx.x;          // 0..127, == output column
    xs[t] = x[(size_t)node * IN_DIM + t];
    __syncthreads();

    float acc = 0.0f;
#pragma unroll 8
    for (int k = 0; k < IN_DIM; ++k)
        acc = fmaf(xs[k], W[(size_t)k * OC + t], acc);

    const int d = t & 31;                  // dim within head; wave == head
    float s = acc * a[d];                  // a_j = a[0..31]
#pragma unroll
    for (int off = 16; off > 0; off >>= 1)
        s += __shfl_xor(s, off, 32);
    const float u = __expf(s * 0.17677669529663687f);  // scale = 1/sqrt(32)

    Vt[(size_t)t * NN + node] = (_Float16)(u * acc);
    if (d == 0)
        Vt[(size_t)(128 + (t >> 5)) * NN + node] = (_Float16)u;   // denom cols
    if (t < 12)
        Vt[(size_t)(132 + t) * NN + node] = (_Float16)0.0f;       // zero pad cols
}

// ---------------------------------------------------------------------------
// Kernel 2: Raw = adj @ V via v_wmma_f32_16x16x32_f16, double-buffered async
// LDS staging, fused softmax normalization.
// Block: 64 threads = 2 waves; each wave owns a 16-row M-tile and all 9
// N-tiles (72 accumulator VGPRs).
// ---------------------------------------------------------------------------
__global__ void __launch_bounds__(64)
gat_attn_wmma_kernel(const float* __restrict__ adj,
                     const _Float16* __restrict__ Vt,
                     float* __restrict__ out) {
    __shared__ __align__(16) _Float16 vT[2][NCOL][VPAD];  // [buf][col][k]
    __shared__ float denomLds[BM][4];

    const int t       = threadIdx.x;      // 0..63
    const int lane    = t & 31;
    const int wave    = t >> 5;           // 0..1
    const int rowBase = blockIdx.x * BM;
    const int mBase   = rowBase + wave * 16;
    const int mRow    = mBase + (lane & 15);          // A-frag row for this lane
    const int kloA    = (lane < 16) ? 0 : 8;          // A-frag K sub-block
    const int kloB    = (lane < 16) ? 0 : 8;          // B-frag K sub-block
    const int colB    = lane & 15;                    // B/C col within tile

    // Per-thread async-copy slots: idx = t + i*64, c = idx>>2, q = idx&3
    const unsigned ldsBase0 = (unsigned)(uintptr_t)&vT[0][0][0];
    const unsigned ldsBase1 = (unsigned)(uintptr_t)&vT[1][0][0];

    v8f acc[NT] = {};                                 // 9 x 16x16 f32 tiles
    const float* adjRow = adj + (size_t)mRow * NN;

    // ---- prologue: stage K-block 0 into buffer 0 (9 async ops per wave)
#pragma unroll
    for (int i = 0; i < 9; ++i) {
        const int idx = t + i * 64;
        const int c = idx >> 2, q = idx & 3;
        async_cp16(ldsBase0 + (unsigned)(c * (VPAD * 2) + q * 16),
                   (const void*)(Vt + (size_t)c * NN + q * 8));
    }

    for (int kb = 0; kb < NN; kb += BK) {
        const int  cur     = (kb >> 5) & 1;
        const bool hasNext = (kb + BK) < NN;

        // ---- issue async staging for the NEXT K-block into the other buffer
        if (hasNext) {
            const unsigned base = cur ? ldsBase0 : ldsBase1;
#pragma unroll
            for (int i = 0; i < 9; ++i) {
                const int idx = t + i * 64;
                const int c = idx >> 2, q = idx & 3;
                async_cp16(base + (unsigned)(c * (VPAD * 2) + q * 16),
                           (const void*)(Vt + (size_t)c * NN + (kb + BK) + q * 8));
            }
            wait_async_le9();   // drain CURRENT group; next stays in flight
        } else {
            wait_async_le0();
        }
        __syncthreads();        // staged data visible block-wide

        // ---- A fragment: adj[mRow][kb+kloA..+7] and [kb+kloA+16..+23] (f32->f16)
        const v4f a0 = *(const v4f*)(adjRow + kb + kloA);
        const v4f a1 = *(const v4f*)(adjRow + kb + kloA + 4);
        const v4f a2 = *(const v4f*)(adjRow + kb + kloA + 16);
        const v4f a3 = *(const v4f*)(adjRow + kb + kloA + 20);
        v16h af;
#pragma unroll
        for (int i = 0; i < 4; ++i) {
            af[i]      = (_Float16)a0[i];
            af[4 + i]  = (_Float16)a1[i];
            af[8 + i]  = (_Float16)a2[i];
            af[12 + i] = (_Float16)a3[i];
        }

        // ---- preload all 9 B fragments, then 9 back-to-back WMMAs
        v16h bf[NT];
#pragma unroll
        for (int nt = 0; nt < NT; ++nt) {
            const v8h b0 = *(const v8h*)(const void*)(&vT[cur][nt * 16 + colB][kloB]);
            const v8h b1 = *(const v8h*)(const void*)(&vT[cur][nt * 16 + colB][kloB + 16]);
#pragma unroll
            for (int i = 0; i < 8; ++i) { bf[nt][i] = b0[i]; bf[nt][8 + i] = b1[i]; }
        }
#pragma unroll
        for (int nt = 0; nt < NT; ++nt)
            acc[nt] = __builtin_amdgcn_wmma_f32_16x16x32_f16(
                false, af, false, bf[nt], (short)0, acc[nt], false, false);

        __syncthreads();        // all waves done reading cur before it is reused
    }

    // ---- epilogue: tile 8 local cols 0..3 hold per-head denominators
    const int rshift = (lane >> 4) * 8;               // C-layout row offset
    if (colB < 4) {
#pragma unroll
        for (int r = 0; r < 8; ++r)
            denomLds[wave * 16 + r + rshift][colB] = acc[8][r];
    }
    __syncthreads();

#pragma unroll
    for (int nt = 0; nt < 8; ++nt) {
        const int c = nt * 16 + colB;
        const int h = c >> 5;
#pragma unroll
        for (int r = 0; r < 8; ++r) {
            const int mloc = wave * 16 + r + rshift;
            out[(size_t)(rowBase + mloc) * OC + c] = acc[nt][r] / denomLds[mloc][h];
        }
    }
}

// ---------------------------------------------------------------------------
extern "C" void kernel_launch(void* const* d_in, const int* in_sizes, int n_in,
                              void* d_out, int out_size, void* d_ws, size_t ws_size,
                              hipStream_t stream) {
    const float* x   = (const float*)d_in[0];   // (4096,128)
    const float* adj = (const float*)d_in[1];   // (4096,4096)
    const float* W   = (const float*)d_in[2];   // (128,128)
    const float* a   = (const float*)d_in[3];   // (64,)
    float* out = (float*)d_out;                 // (4096,128)

    _Float16* Vt = (_Float16*)d_ws;             // 144 * 4096 * 2B = 1.18 MB

    gat_xp_v_kernel<<<NN, 128, 0, stream>>>(x, W, a, Vt);
    gat_attn_wmma_kernel<<<NN / BM, 64, 0, stream>>>(adj, Vt, out);
}
